// MultiHeadAttention_10703058501687
// MI455X (gfx1250) — matrix operand order repaired and checked by reading, not compiled, not run on hardware
//
#include <hip/hip_runtime.h>

// ---------------------------------------------------------------------------
// CDNA5 (gfx1250) multi-head attention, bf16 WMMA path, wave32.
// ---------------------------------------------------------------------------

typedef __attribute__((ext_vector_type(16))) __bf16 v16bf;
typedef __attribute__((ext_vector_type(8)))  __bf16 v8bf;
typedef __attribute__((ext_vector_type(4)))  __bf16 v4bf;
typedef __attribute__((ext_vector_type(8)))  float  v8f;

#define D_MODEL 1024
#define N_HEAD  16
#define DEPTH   64
#define BATCH   4
#define SEQ     2048
#define ROWS    (BATCH * SEQ) // 8192

__device__ __forceinline__ __bf16 f2bf(float f) {
  unsigned u = __builtin_bit_cast(unsigned, f);
  unsigned r = u + 0x7FFFu + ((u >> 16) & 1u); // round-to-nearest-even
  return __builtin_bit_cast(__bf16, (unsigned short)(r >> 16));
}

__device__ __forceinline__ v8f wmma_bf16(v16bf a, v16bf b, v8f c) {
  // D = A(16x32 bf16) * B(32x16 bf16) + C(16x16 f32)
  return __builtin_amdgcn_wmma_f32_16x16x32_bf16(
      /*neg_a=*/false, a, /*neg_b=*/false, b,
      /*c_mod=*/(short)0, c, /*reuse_a=*/false, /*reuse_b=*/false);
}

// A-matrix fragment (16x32 bf16): lane = row, 16 contiguous K chosen by half
__device__ __forceinline__ v16bf loadA(const __bf16* row, int kk, int half) {
  v8bf lo = *(const v8bf*)(row + kk + half * 16);
  v8bf hi = *(const v8bf*)(row + kk + half * 16 + 8);
  v16bf a;
#pragma unroll
  for (int i = 0; i < 8; ++i) { a[i] = lo[i]; a[8 + i] = hi[i]; }
  return a;
}

// ---------------------------------------------------------------------------
// fp32 -> bf16 bulk conversion (bandwidth-bound, one pass)
// ---------------------------------------------------------------------------
__global__ __launch_bounds__(256) void cvt_bf16_kernel(
    const float* __restrict__ in, __bf16* __restrict__ out, int n4)
{
  int i = blockIdx.x * blockDim.x + threadIdx.x;
  if (i >= n4) return;
  const float4 f = ((const float4*)in)[i];
  v4bf o;
  o[0] = f2bf(f.x); o[1] = f2bf(f.y); o[2] = f2bf(f.z); o[3] = f2bf(f.w);
  ((v4bf*)out)[i] = o;
}

// ---------------------------------------------------------------------------
// out[m,n] = sum_k X[m,k] * W[n,k] + bias[n]   (X: [8192,1024], W: [1024,1024])
// One wave computes a 16x64 output tile (1 A-frag x 4 B-frags, 4 accumulators).
// K-loop unrolled x2 with ping-pong fragment buffers: load stage1, WMMA
// stage0, load next stage0, WMMA stage1 -> staggered s_wait_loadcnt.
// __launch_bounds__(256, 1) lifts the allocator's VGPR budget so prefetched
// fragments get their own registers instead of recycling accumulator ranges
// (which forced acc v_mov copies + WMMA hazard nops in earlier builds).
// mode 0: fp32 store, plain [8192,1024]                     (final output)
// mode 1: bf16 store, head-split  [B,H,S,DEPTH]             (q, k)
// mode 2: bf16 store, head-split transposed [B,H,DEPTH,S]   (v)
// ---------------------------------------------------------------------------
__global__ __launch_bounds__(256, 1) void gemm_bt_kernel(
    const __bf16* __restrict__ X, const __bf16* __restrict__ W,
    const float* __restrict__ bias, void* __restrict__ outv, int mode)
{
  const int lane   = threadIdx.x & 31;
  const int wave   = threadIdx.x >> 5;
  const int gw     = blockIdx.x * 8 + wave;    // 8192 waves total
  const int tileN  = (gw & 15) * 64;           // 16 column tiles of 64
  const int m0     = (gw >> 4) * 16;           // 512 row tiles of 16
  const int lanelo = lane & 15;
  const int half   = lane >> 4;

  const __bf16* xrow = X + (size_t)(m0 + lanelo) * D_MODEL;
  const __bf16* wcol[4];
#pragma unroll
  for (int t = 0; t < 4; ++t)
    wcol[t] = W + (size_t)(tileN + t * 16 + lanelo) * D_MODEL + half * 16;

  v8f acc[4] = {};
  v16bf Af[2], Bf[2][4];

  // prologue: stage 0 = kk 0
  Af[0] = loadA(xrow, 0, half);
#pragma unroll
  for (int t = 0; t < 4; ++t) Bf[0][t] = *(const v16bf*)(wcol[t]);

  int kk = 0;
  for (; kk < D_MODEL - 64; kk += 64) {
    // load stage 1 (kk+32), then consume stage 0
    Af[1] = loadA(xrow, kk + 32, half);
#pragma unroll
    for (int t = 0; t < 4; ++t) Bf[1][t] = *(const v16bf*)(wcol[t] + kk + 32);
#pragma unroll
    for (int t = 0; t < 4; ++t) acc[t] = wmma_bf16(Af[0], Bf[0][t], acc[t]);

    // load next stage 0 (kk+64), then consume stage 1
    Af[0] = loadA(xrow, kk + 64, half);
#pragma unroll
    for (int t = 0; t < 4; ++t) Bf[0][t] = *(const v16bf*)(wcol[t] + kk + 64);
#pragma unroll
    for (int t = 0; t < 4; ++t) acc[t] = wmma_bf16(Af[1], Bf[1][t], acc[t]);
  }
  // epilogue: kk = D_MODEL-64; stage 0 already loaded, stage 1 = last 32
  Af[1] = loadA(xrow, kk + 32, half);
#pragma unroll
  for (int t = 0; t < 4; ++t) Bf[1][t] = *(const v16bf*)(wcol[t] + kk + 32);
#pragma unroll
  for (int t = 0; t < 4; ++t) acc[t] = wmma_bf16(Af[0], Bf[0][t], acc[t]);
#pragma unroll
  for (int t = 0; t < 4; ++t) acc[t] = wmma_bf16(Af[1], Bf[1][t], acc[t]);

#pragma unroll
  for (int t = 0; t < 4; ++t) {
    const int col = tileN + t * 16 + lanelo;
    const float bv = bias[col];
#pragma unroll
    for (int j = 0; j < 8; ++j) {
      const int m = m0 + j + 8 * half;           // C layout: vgpr j, +8 hi half
      const float val = acc[t][j] + bv;
      if (mode == 0) {
        ((float*)outv)[(size_t)m * D_MODEL + col] = val;
      } else {
        const int bb = m >> 11, s = m & (SEQ - 1);
        const int h = col >> 6, d = col & (DEPTH - 1);
        const size_t dst = (mode == 1)
            ? ((size_t)(bb * N_HEAD + h) * SEQ + s) * DEPTH + d
            : ((size_t)(bb * N_HEAD + h) * DEPTH + d) * SEQ + s;
        ((__bf16*)outv)[dst] = f2bf(val);
      }
    }
  }
}

// ---------------------------------------------------------------------------
// Causal flash attention. One wave owns a 16-query tile of one (b,h).
// q,k: [B,H,S,64] bf16    vt: [B,H,64,S] bf16    x out: [8192,1024] bf16
// ---------------------------------------------------------------------------
__global__ __launch_bounds__(256) void attn_kernel(
    const __bf16* __restrict__ q, const __bf16* __restrict__ k,
    const __bf16* __restrict__ vt, __bf16* __restrict__ x)
{
  __shared__ __bf16 pbuf[8][16][48]; // per-wave P staging (C-layout -> A-layout)

  const int lane   = threadIdx.x & 31;
  const int wave   = threadIdx.x >> 5;
  const int gw     = blockIdx.x * 8 + wave;  // 8192 waves = B*H*S/16 q-tiles
  const int bh     = gw >> 7;                // b*16 + h
  const int q0     = (gw & 127) * 16;
  const int lanelo = lane & 15;
  const int half   = lane >> 4;

  const __bf16* qp = q  + (size_t)bh * SEQ * DEPTH;
  const __bf16* kp = k  + (size_t)bh * SEQ * DEPTH;
  const __bf16* vp = vt + (size_t)bh * DEPTH * SEQ;

  // Q A-fragments: depth 0..31 and 32..63
  const __bf16* qrow = qp + (size_t)(q0 + lanelo) * DEPTH;
  v16bf aq[2];
#pragma unroll
  for (int h2 = 0; h2 < 2; ++h2) aq[h2] = loadA(qrow, h2 * 32, half);

  v8f acc[4] = {};
  float mrun[8], lrun[8];
#pragma unroll
  for (int j = 0; j < 8; ++j) { mrun[j] = -3.0e38f; lrun[j] = 0.0f; }
  const float scale = 0.125f; // 1/sqrt(DEPTH)
  const int nk = q0 + 16;     // causal: keys 0 .. q0+15

  for (int ks = 0; ks < nk; ks += 32) {
    // scores: two 16x16 tiles (keys ks..+15, ks+16..+31), depth K=64 via 2 WMMAs
    v8f sc[2] = {};
#pragma unroll
    for (int t = 0; t < 2; ++t) {
      const __bf16* krow = kp + (size_t)(ks + t * 16 + lanelo) * DEPTH + half * 16;
      sc[t] = wmma_bf16(aq[0], *(const v16bf*)(krow), sc[t]);
      sc[t] = wmma_bf16(aq[1], *(const v16bf*)(krow + 32), sc[t]);
    }

    // V fragments for the PV WMMAs: independent of the softmax below, issue
    // them now so their latency hides behind the shfl/exp VALU chain.
    v16bf bv[4];
#pragma unroll
    for (int t = 0; t < 4; ++t) {
      const __bf16* vrow = vp + (size_t)(t * 16 + lanelo) * SEQ + ks + half * 16;
      bv[t] = *(const v16bf*)vrow;
    }
    __builtin_prefetch(kp + (size_t)(ks + 32 + lanelo) * DEPTH, 0, 0);

    // online softmax: mask, row-max, rescale running state, write P to LDS
#pragma unroll
    for (int j = 0; j < 8; ++j) {
      const int qidx = q0 + j + 8 * half;
      float s0 = (ks + lanelo      > qidx) ? -1.0e9f : sc[0][j] * scale;
      float s1 = (ks + 16 + lanelo > qidx) ? -1.0e9f : sc[1][j] * scale;
      float vmax = fmaxf(s0, s1);
#pragma unroll
      for (int off = 8; off >= 1; off >>= 1)
        vmax = fmaxf(vmax, __shfl_xor(vmax, off, 32)); // stays inside 16-lane half
      const float mnew = fmaxf(mrun[j], vmax);
      const float corr = __expf(mrun[j] - mnew);
      const float p0 = __expf(s0 - mnew);
      const float p1 = __expf(s1 - mnew);
      const int row = j + 8 * half;
      pbuf[wave][row][lanelo]      = f2bf(p0);
      pbuf[wave][row][16 + lanelo] = f2bf(p1);
      float psum = p0 + p1;
#pragma unroll
      for (int off = 8; off >= 1; off >>= 1)
        psum += __shfl_xor(psum, off, 32);
      lrun[j] = lrun[j] * corr + psum;
      mrun[j] = mnew;
#pragma unroll
      for (int t = 0; t < 4; ++t) acc[t][j] *= corr;
    }

    __builtin_amdgcn_wave_barrier(); // per-wave LDS, DS ops in-order within wave

    // reload P as A-fragment (16 queries x 32 keys)
    v16bf pf;
    {
      v8bf lo = *(const v8bf*)&pbuf[wave][lanelo][half * 16];
      v8bf hi = *(const v8bf*)&pbuf[wave][lanelo][half * 16 + 8];
#pragma unroll
      for (int i = 0; i < 8; ++i) { pf[i] = lo[i]; pf[8 + i] = hi[i]; }
    }

    // acc += P(16x32) @ V(32x64)
#pragma unroll
    for (int t = 0; t < 4; ++t) acc[t] = wmma_bf16(pf, bv[t], acc[t]);
  }

  // finalize: divide by l, merge heads into x[b, s, h*64 + d]
  const int h  = bh & (N_HEAD - 1);
  const int bb = bh >> 4;
#pragma unroll
  for (int j = 0; j < 8; ++j) {
    const float inv = 1.0f / lrun[j];
    const int m = q0 + j + 8 * half;
#pragma unroll
    for (int t = 0; t < 4; ++t) {
      const int col = h * DEPTH + t * 16 + lanelo;
      x[((size_t)(bb * SEQ + m)) * D_MODEL + col] = f2bf(acc[t][j] * inv);
    }
  }
}

// ---------------------------------------------------------------------------
// Host-side orchestration
// ---------------------------------------------------------------------------
extern "C" void kernel_launch(void* const* d_in, const int* in_sizes, int n_in,
                              void* d_out, int out_size, void* d_ws, size_t ws_size,
                              hipStream_t stream) {
  (void)in_sizes; (void)n_in; (void)out_size; (void)ws_size;
  const float* Q  = (const float*)d_in[0];
  const float* K  = (const float*)d_in[1];
  const float* V  = (const float*)d_in[2];
  /* d_in[3] = causal mask: applied analytically in attn_kernel */
  const float* Wq = (const float*)d_in[4];
  const float* bq = (const float*)d_in[5];
  const float* Wk = (const float*)d_in[6];
  const float* bk = (const float*)d_in[7];
  const float* Wv = (const float*)d_in[8];
  const float* bv = (const float*)d_in[9];
  const float* Wo = (const float*)d_in[10];
  const float* bo = (const float*)d_in[11];

  const size_t NA = (size_t)ROWS * D_MODEL;      // 8M activation elems
  const size_t NW = (size_t)D_MODEL * D_MODEL;   // 1M weight elems

  __bf16* p   = (__bf16*)d_ws;
  __bf16* Qb  = p; p += NA;
  __bf16* Kb  = p; p += NA;
  __bf16* Vb  = p; p += NA;
  __bf16* Wqb = p; p += NW;
  __bf16* Wkb = p; p += NW;
  __bf16* Wvb = p; p += NW;
  __bf16* Wob = p; p += NW;
  __bf16* qb  = p; p += NA;  // [B,H,S,64]
  __bf16* kb  = p; p += NA;  // [B,H,S,64]
  __bf16* vtb = p; p += NA;  // [B,H,64,S]
  __bf16* xb  = p; p += NA;  // [8192,1024]

  const dim3 blk(256);
  const int gA = (int)(NA / 4 / 256);  // 8192 blocks
  const int gW = (int)(NW / 4 / 256);  // 1024 blocks

  cvt_bf16_kernel<<<gA, blk, 0, stream>>>(Q,  Qb,  (int)(NA / 4));
  cvt_bf16_kernel<<<gA, blk, 0, stream>>>(K,  Kb,  (int)(NA / 4));
  cvt_bf16_kernel<<<gA, blk, 0, stream>>>(V,  Vb,  (int)(NA / 4));
  cvt_bf16_kernel<<<gW, blk, 0, stream>>>(Wq, Wqb, (int)(NW / 4));
  cvt_bf16_kernel<<<gW, blk, 0, stream>>>(Wk, Wkb, (int)(NW / 4));
  cvt_bf16_kernel<<<gW, blk, 0, stream>>>(Wv, Wvb, (int)(NW / 4));
  cvt_bf16_kernel<<<gW, blk, 0, stream>>>(Wo, Wob, (int)(NW / 4));

  gemm_bt_kernel<<<1024, blk, 0, stream>>>(Qb, Wqb, bq, qb,  1);
  gemm_bt_kernel<<<1024, blk, 0, stream>>>(Kb, Wkb, bk, kb,  1);
  gemm_bt_kernel<<<1024, blk, 0, stream>>>(Vb, Wvb, bv, vtb, 2);

  attn_kernel<<<1024, blk, 0, stream>>>(qb, kb, vtb, xb);

  gemm_bt_kernel<<<1024, blk, 0, stream>>>(xb, Wob, bo, d_out, 0);
}
